// DichotomyIR_24077586661379
// MI455X (gfx1250) — matrix operand order repaired and checked by reading, not compiled, not run on hardware
//
#include <hip/hip_runtime.h>
#include <hip/hip_fp16.h>

// MI455X / gfx1250 implementation.
// WMMA (v_wmma_f32_16x16x32_f16) for all channel contractions (in_proj, x_dbl,
// out_proj) — GEMM dims are template constants so fragment register arrays are
// statically indexed (no v_movrels). Chunked associative scan for the SS2D
// selective-scan (serial depth 4096 -> 64+64); gather-based direction
// flips/transposes (no materialized xs).

typedef __attribute__((ext_vector_type(16))) _Float16 v16h;
typedef __attribute__((ext_vector_type(8)))  float    v8f;

#define LQ      4096   // H*W
#define HW      64
#define DINNER  48
#define NSTATE  16
#define CBIN    24
#define NBAT    4
#define KDIR    4
#define CH_LEN  64     // scan chunk length
#define N_CHUNK 64     // chunks per chain (CH_LEN*N_CHUNK == LQ)

// -------- direction index maps (involutions for k=0,1; pairs for 2,3) --------
__device__ __forceinline__ int map_dir(int k, int l) {
  switch (k & 3) {
    case 0:  return l;
    case 1:  return ((l & 63) << 6) | (l >> 6);           // (h,w)<->(w,h)
    case 2:  return LQ - 1 - l;
    default: { int m = LQ - 1 - l; return ((m & 63) << 6) | (m >> 6); }
  }
}

__device__ __forceinline__ float softplusf(float x) {
  return (x > 20.f) ? x : log1pf(__expf(x));
}
__device__ __forceinline__ float sigmoidf(float x) { return 1.f / (1.f + __expf(-x)); }

// ------------------------------ bit planes -----------------------------------
__global__ void k_dec2bin(const float* __restrict__ img, float* __restrict__ bins) {
  int i = blockIdx.x * blockDim.x + threadIdx.x;          // [3][B][24][L]
  if (i >= 3 * NBAT * CBIN * LQ) return;
  int p   = i & (LQ - 1);
  int t   = i >> 12;
  int c24 = t % CBIN;   t /= CBIN;
  int b   = t % NBAT;
  int br  = t / NBAT;
  int c = c24 >> 3, j = c24 & 7;
  float x = img[(((size_t)br * NBAT + b) * 3 + c) * LQ + p];
  unsigned v = (unsigned)(x * 255.0f);
  bins[i] = (float)((v >> j) & 1u);
}

__global__ void k_init(const float* __restrict__ bins, float* __restrict__ refined,
                       float* __restrict__ biasb) {
  int i = blockIdx.x * blockDim.x + threadIdx.x;
  if (i >= NBAT * CBIN * LQ) return;
  float b0 = bins[i];
  refined[i] = b0;
  biasb[i]   = bins[(size_t)NBAT * CBIN * LQ + i] - b0;
}

// ------------- per-iteration prep: LN + log_softmax KL + LN(bias) ------------
__global__ void k_prep(const float* __restrict__ binsI, const float* __restrict__ refined,
                       const float* __restrict__ biasT,
                       const float* __restrict__ gi, const float* __restrict__ bi,
                       const float* __restrict__ gb, const float* __restrict__ bb,
                       float* __restrict__ kld, float* __restrict__ bnout) {
  int i = blockIdx.x * blockDim.x + threadIdx.x;          // [B][L] pixels
  if (i >= NBAT * LQ) return;
  int b = i >> 12, p = i & (LQ - 1);
  const float* P  = binsI   + (size_t)b * CBIN * LQ + p;
  const float* C  = refined + (size_t)b * CBIN * LQ + p;
  const float* Bs = biasT   + (size_t)b * CBIN * LQ + p;
  float muP = 0, sP = 0, muC = 0, sC = 0, muB = 0, sB = 0;
  for (int c = 0; c < CBIN; c++) {
    float x = P[c * LQ];  muP += x; sP += x * x;
    float y = C[c * LQ];  muC += y; sC += y * y;
    float z = Bs[c * LQ]; muB += z; sB += z * z;
  }
  muP /= CBIN; muC /= CBIN; muB /= CBIN;
  float rP = rsqrtf(sP / CBIN - muP * muP + 1e-5f);
  float rC = rsqrtf(sC / CBIN - muC * muC + 1e-5f);
  float rB = rsqrtf(sB / CBIN - muB * muB + 1e-5f);
  // stable log-softmax over channels for prev (P) and cur (C)
  float mP = -1e30f, mC = -1e30f;
  for (int c = 0; c < CBIN; c++) {
    float lnp = (P[c * LQ] - muP) * rP * gi[c] + bi[c];
    float lnc = (C[c * LQ] - muC) * rC * gi[c] + bi[c];
    mP = fmaxf(mP, lnp); mC = fmaxf(mC, lnc);
  }
  float eP = 0.f, eC = 0.f;
  for (int c = 0; c < CBIN; c++) {
    float lnp = (P[c * LQ] - muP) * rP * gi[c] + bi[c];
    float lnc = (C[c * LQ] - muC) * rC * gi[c] + bi[c];
    eP += __expf(lnp - mP); eC += __expf(lnc - mC);
  }
  float lseP = mP + logf(eP), lseC = mC + logf(eC);
  for (int c = 0; c < CBIN; c++) {
    float lnp = (P[c * LQ] - muP) * rP * gi[c] + bi[c];
    float lnc = (C[c * LQ] - muC) * rC * gi[c] + bi[c];
    float lq = lnp - lseP, lp = lnc - lseC;
    kld  [(size_t)b * CBIN * LQ + c * LQ + p] = __expf(lq) * (lq - lp);
    bnout[(size_t)b * CBIN * LQ + c * LQ + p] = (Bs[c * LQ] - muB) * rB * gb[c] + bb[c];
  }
}

// ------------- generic WMMA GEMM: Y[b,n,p] = sum_c X[b,c,p] W[n,c] + bias ----
// K, N compile-time so all fragment loops fully unroll (no movrels).
// grid = (LQ/64, NBAT), block = 128 (4 waves, 16 M-rows each)
template <int K, int N>
__global__ void k_gemm(const float* __restrict__ X, const float* __restrict__ W,
                       const float* __restrict__ bias, float* __restrict__ Y) {
  constexpr int Kpad   = (K + 31) & ~31;
  constexpr int Ntiles = (N + 15) >> 4;
  constexpr int Npad   = Ntiles << 4;
  constexpr int NKT    = Kpad >> 5;                       // 1 or 2 K-steps
  __shared__ _Float16 At[64 * Kpad];                      // [m][Kpad]
  __shared__ _Float16 Bt[Npad * Kpad];                    // [n][Kpad]
  int b  = blockIdx.y;
  int p0 = blockIdx.x * 64;
  int tid = threadIdx.x;
  // A staging: position index fastest -> coalesced global reads
  for (int i = tid; i < 64 * Kpad; i += blockDim.x) {
    int c = i >> 6, m = i & 63;
    At[m * Kpad + c] =
        (c < K) ? (_Float16)X[((size_t)b * K + c) * LQ + p0 + m] : (_Float16)0.f;
  }
  // B staging: channel index fastest -> coalesced reads of row-major W[n][c]
  for (int i = tid; i < Npad * Kpad; i += blockDim.x) {
    int n = i / Kpad, c = i & (Kpad - 1);
    Bt[i] = (n < N && c < K) ? (_Float16)W[n * K + c] : (_Float16)0.f;
  }
  __syncthreads();
  int lane = tid & 31, wave = tid >> 5;
  int m0 = wave * 16;
  int hi = (lane >> 4) & 1, lm = lane & 15;
  v16h afr[NKT];
#pragma unroll
  for (int kt = 0; kt < NKT; kt++)
#pragma unroll
    for (int e = 0; e < 16; e++) {
      int kk = kt * 32 + ((e >> 3) << 4) + (hi << 3) + (e & 7);
      afr[kt][e] = At[(m0 + lm) * Kpad + kk];
    }
#pragma unroll
  for (int nt = 0; nt < Ntiles; nt++) {
    v8f acc = {0.f, 0.f, 0.f, 0.f, 0.f, 0.f, 0.f, 0.f};
#pragma unroll
    for (int kt = 0; kt < NKT; kt++) {
      v16h bfr;
#pragma unroll
      for (int e = 0; e < 16; e++) {
        int kk = kt * 32 + ((e >> 3) << 4) + (hi << 3) + (e & 7);
        bfr[e] = Bt[(nt * 16 + lm) * Kpad + kk];
      }
      acc = __builtin_amdgcn_wmma_f32_16x16x32_f16(false, afr[kt], false, bfr,
                                                   (short)0, acc, false, false);
    }
#pragma unroll
    for (int r = 0; r < 8; r++) {
      int M = r + (hi << 3);
      int n = nt * 16 + lm;
      if (n < N)
        Y[((size_t)b * N + n) * LQ + p0 + m0 + M] = acc[r] + (bias ? bias[n] : 0.f);
    }
  }
}

// ------------------ depthwise 3x3 conv + bias + SiLU on xi half --------------
__global__ void k_dwconv(const float* __restrict__ xz, const float* __restrict__ cw,
                         const float* __restrict__ cb, float* __restrict__ xconv) {
  int i = blockIdx.x * blockDim.x + threadIdx.x;          // [B][48][L]
  if (i >= NBAT * DINNER * LQ) return;
  int p = i & (LQ - 1);
  int t = i >> 12;
  int d = t % DINNER, b = t / DINNER;
  int h = p >> 6, w = p & 63;
  const float* in = xz + ((size_t)b * 96 + d) * LQ;
  const float* wt = cw + d * 9;
  float s = cb[d];
  for (int kh = 0; kh < 3; kh++) {
    int hh = h + kh - 1; if (hh < 0 || hh > 63) continue;
    for (int kw = 0; kw < 3; kw++) {
      int ww = w + kw - 1; if (ww < 0 || ww > 63) continue;
      s += in[hh * HW + ww] * wt[kh * 3 + kw];
    }
  }
  xconv[((size_t)b * DINNER + d) * LQ + p] = s * sigmoidf(s);
}

// -------- x_dbl: per-direction WMMA GEMM with gathered (flip/transposed) A ---
// grid = (64, NBAT, KDIR), block = 128
__global__ void k_xdbl(const float* __restrict__ xconv, const float* __restrict__ xproj,
                       float* __restrict__ xdbl) {
  constexpr int K = DINNER, Kpad = 64, N = 34, Ntiles = 3;
  __shared__ _Float16 At[64 * Kpad];   // [m][64]
  __shared__ _Float16 Bt[48 * Kpad];   // [n][64]
  int b = blockIdx.y, k = blockIdx.z;
  int p0 = blockIdx.x * 64;
  int tid = threadIdx.x;
  // position fastest for (mostly) coalesced gathers
  for (int i = tid; i < 64 * Kpad; i += blockDim.x) {
    int c = i >> 6, m = i & 63;
    float v = 0.f;
    if (c < K) v = xconv[((size_t)b * K + c) * LQ + map_dir(k, p0 + m)];
    At[m * Kpad + c] = (_Float16)v;
  }
  for (int i = tid; i < 48 * Kpad; i += blockDim.x) {
    int n = i >> 6, c = i & 63;
    Bt[i] = (n < N && c < K) ? (_Float16)xproj[(k * 34 + n) * 48 + c] : (_Float16)0.f;
  }
  __syncthreads();
  int lane = tid & 31, wave = tid >> 5;
  int m0 = wave * 16;
  int hi = (lane >> 4) & 1, lm = lane & 15;
  v16h afr[2];
#pragma unroll
  for (int kt = 0; kt < 2; kt++)
#pragma unroll
    for (int e = 0; e < 16; e++) {
      int kk = kt * 32 + ((e >> 3) << 4) + (hi << 3) + (e & 7);
      afr[kt][e] = At[(m0 + lm) * Kpad + kk];
    }
#pragma unroll
  for (int nt = 0; nt < Ntiles; nt++) {
    v8f acc = {0.f, 0.f, 0.f, 0.f, 0.f, 0.f, 0.f, 0.f};
#pragma unroll
    for (int kt = 0; kt < 2; kt++) {
      v16h bfr;
#pragma unroll
      for (int e = 0; e < 16; e++) {
        int kk = kt * 32 + ((e >> 3) << 4) + (hi << 3) + (e & 7);
        bfr[e] = Bt[(nt * 16 + lm) * Kpad + kk];
      }
      acc = __builtin_amdgcn_wmma_f32_16x16x32_f16(false, afr[kt], false, bfr,
                                                   (short)0, acc, false, false);
    }
#pragma unroll
    for (int r = 0; r < 8; r++) {
      int M = r + (hi << 3);
      int n = nt * 16 + lm;
      if (n < N)
        xdbl[(((size_t)b * KDIR + k) * 34 + n) * LQ + p0 + m0 + M] = acc[r];
    }
  }
}

// ------------------------------ dt = softplus(...) ---------------------------
__global__ void k_dts(const float* __restrict__ xdbl, const float* __restrict__ dtw,
                      const float* __restrict__ dtb, float* __restrict__ dts) {
  int i = blockIdx.x * blockDim.x + threadIdx.x;          // [B][K][D][L]
  if (i >= NBAT * KDIR * DINNER * LQ) return;
  int p = i & (LQ - 1);
  int t = i >> 12;
  int d = t % DINNER; t /= DINNER;
  int k = t & 3;      int b = t >> 2;
  const float* base = xdbl + (((size_t)b * KDIR + k) * 34) * LQ;
  float x = base[p] * dtw[(k * DINNER + d) * 2] +
            base[LQ + p] * dtw[(k * DINNER + d) * 2 + 1] + dtb[k * DINNER + d];
  dts[i] = softplusf(x);
}

// --------- chunked selective scan: pass 1 (local chunk scan, dt sums) --------
// 16 lanes per (chain,chunk) work unit; block 256 = 16 units
__global__ void k_scan_p1(const float* __restrict__ dts, const float* __restrict__ xdbl,
                          const float* __restrict__ xconv, const float* __restrict__ alog,
                          float* __restrict__ hend, float* __restrict__ sdt) {
  int unit = blockIdx.x * 16 + (threadIdx.x >> 4);
  int n = threadIdx.x & 15;
  int chain = unit >> 6, chunk = unit & 63;
  int d = chain % DINNER; int t = chain / DINNER;
  int k = t & 3, b = t >> 2;
  float A = -__expf(alog[(k * DINNER + d) * NSTATE + n]);
  const float* dtp = dts + (size_t)chain * LQ;
  const float* Bp  = xdbl  + (((size_t)b * KDIR + k) * 34 + 2 + n) * LQ;
  const float* xp  = xconv + ((size_t)b * DINNER + d) * LQ;
  float h = 0.f, s = 0.f;
  int l0 = chunk * CH_LEN;
  for (int j = 0; j < CH_LEN; j++) {
    int l = l0 + j;
    float dt = dtp[l];
    float xv = xp[map_dir(k, l)];
    h = __expf(dt * A) * h + dt * xv * Bp[l];
    s += dt;
  }
  hend[(size_t)unit * NSTATE + n] = h;
  if (n == 0) sdt[unit] = s;
}

// --------------- pass 2 (mid): sequential combine over 64 chunks -------------
__global__ void k_scan_mid(const float* __restrict__ hend, const float* __restrict__ sdt,
                           const float* __restrict__ alog, float* __restrict__ h0) {
  int idx = blockIdx.x * blockDim.x + threadIdx.x;        // [chains][16]
  if (idx >= NBAT * KDIR * DINNER * NSTATE) return;
  int n = idx & 15, chain = idx >> 4;
  int d = chain % DINNER; int t = chain / DINNER;
  int k = t & 3;
  float A = -__expf(alog[(k * DINNER + d) * NSTATE + n]);
  float h = 0.f;
  for (int j = 0; j < N_CHUNK; j++) {
    size_t u = (size_t)chain * N_CHUNK + j;
    h0[u * NSTATE + n] = h;
    h = h * __expf(A * sdt[u]) + hend[u * NSTATE + n];
  }
}

// ------ pass 3: replay chunks with true initial state, emit y = <h, C> -------
__global__ void k_scan_p2(const float* __restrict__ dts, const float* __restrict__ xdbl,
                          const float* __restrict__ xconv, const float* __restrict__ alog,
                          const float* __restrict__ h0, float* __restrict__ ys) {
  int unit = blockIdx.x * 16 + (threadIdx.x >> 4);
  int n = threadIdx.x & 15;
  int chain = unit >> 6, chunk = unit & 63;
  int d = chain % DINNER; int t = chain / DINNER;
  int k = t & 3, b = t >> 2;
  float A = -__expf(alog[(k * DINNER + d) * NSTATE + n]);
  const float* dtp = dts + (size_t)chain * LQ;
  const float* Bp  = xdbl  + (((size_t)b * KDIR + k) * 34 + 2 + n) * LQ;
  const float* Cp  = xdbl  + (((size_t)b * KDIR + k) * 34 + 18 + n) * LQ;
  const float* xp  = xconv + ((size_t)b * DINNER + d) * LQ;
  float* yout = ys + (size_t)chain * LQ;
  float h = h0[(size_t)unit * NSTATE + n];
  int l0 = chunk * CH_LEN;
  for (int j = 0; j < CH_LEN; j++) {
    int l = l0 + j;
    float dt = dtp[l];
    float xv = xp[map_dir(k, l)];
    h = __expf(dt * A) * h + dt * xv * Bp[l];
    float c = h * Cp[l];
    c += __shfl_xor(c, 1);
    c += __shfl_xor(c, 2);
    c += __shfl_xor(c, 4);
    c += __shfl_xor(c, 8);
    if (n == 0) yout[l] = c;
  }
}

// ------- combine 4 directions (gather) + D·x skip + LN(D) + SiLU gate --------
__global__ void k_combine(const float* __restrict__ ys, const float* __restrict__ xconv,
                          const float* __restrict__ xz, const float* __restrict__ dpar,
                          const float* __restrict__ ng, const float* __restrict__ nb,
                          float* __restrict__ yg) {
  __shared__ float dsum[DINNER];
  int tid = threadIdx.x;
  if (tid < DINNER)
    dsum[tid] = dpar[tid] + dpar[DINNER + tid] + dpar[2 * DINNER + tid] + dpar[3 * DINNER + tid];
  __syncthreads();
  int i = blockIdx.x * blockDim.x + tid;                  // exactly NBAT*LQ threads
  int b = i >> 12, p = i & (LQ - 1);
  int m1 = ((p & 63) << 6) | (p >> 6);
  const float* y0 = ys + ((size_t)b * KDIR + 0) * DINNER * LQ;
  const float* y1 = ys + ((size_t)b * KDIR + 1) * DINNER * LQ;
  const float* y2 = ys + ((size_t)b * KDIR + 2) * DINNER * LQ;
  const float* y3 = ys + ((size_t)b * KDIR + 3) * DINNER * LQ;
  const float* xc = xconv + (size_t)b * DINNER * LQ;
  float mu = 0.f, sq = 0.f;
  for (int d = 0; d < DINNER; d++) {
    float y = y0[d * LQ + p] + y2[d * LQ + (LQ - 1 - p)] +
              y1[d * LQ + m1] + y3[d * LQ + (LQ - 1 - m1)] + dsum[d] * xc[d * LQ + p];
    mu += y; sq += y * y;
  }
  mu /= DINNER;
  float r = rsqrtf(sq / DINNER - mu * mu + 1e-5f);
  for (int d = 0; d < DINNER; d++) {
    float y = y0[d * LQ + p] + y2[d * LQ + (LQ - 1 - p)] +
              y1[d * LQ + m1] + y3[d * LQ + (LQ - 1 - m1)] + dsum[d] * xc[d * LQ + p];
    float yn = (y - mu) * r * ng[d] + nb[d];
    float z = xz[((size_t)b * 96 + DINNER + d) * LQ + p];
    yg[((size_t)b * DINNER + d) * LQ + p] = yn * z * sigmoidf(z);
  }
}

// ------------------------------ refinement update ----------------------------
__global__ void k_update(float* __restrict__ refined, const float* __restrict__ sigma,
                         const float* __restrict__ bnew) {
  int i = blockIdx.x * blockDim.x + threadIdx.x;
  if (i >= NBAT * CBIN * LQ) return;
  refined[i] = refined[i] * __expf(sigma[i]) + bnew[i];
}

__global__ void k_bin2dec(const float* __restrict__ refined, float* __restrict__ out) {
  int i = blockIdx.x * blockDim.x + threadIdx.x;          // [B][3][L]
  if (i >= NBAT * 3 * LQ) return;
  int p = i & (LQ - 1);
  int t = i >> 12;
  int c = t % 3, b = t / 3;
  float s = 0.f, w = 1.f;
  for (int j = 0; j < 8; j++) { s += refined[((size_t)b * CBIN + c * 8 + j) * LQ + p] * w; w *= 2.f; }
  out[i] = s / 255.f;
}

// =============================== host launcher ===============================
struct SS2DParams {
  const float *in_w, *in_b, *conv_w, *conv_b, *xproj, *dt_w, *dt_b, *alog,
              *dpar, *ng, *nb, *out_w, *out_b;
};

extern "C" void kernel_launch(void* const* d_in, const int* in_sizes, int n_in,
                              void* d_out, int out_size, void* d_ws, size_t ws_size,
                              hipStream_t stream) {
  (void)in_sizes; (void)n_in; (void)out_size; (void)ws_size;
  const float* images = (const float*)d_in[0];
  const float* nimg_g = (const float*)d_in[1];
  const float* nimg_b = (const float*)d_in[2];
  const float* nbias_g = (const float*)d_in[3];
  const float* nbias_b = (const float*)d_in[4];
  auto getp = [&](int base) {
    SS2DParams P;
    P.in_w  = (const float*)d_in[base + 0];  P.in_b  = (const float*)d_in[base + 1];
    P.conv_w= (const float*)d_in[base + 2];  P.conv_b= (const float*)d_in[base + 3];
    P.xproj = (const float*)d_in[base + 4];  P.dt_w  = (const float*)d_in[base + 5];
    P.dt_b  = (const float*)d_in[base + 6];  P.alog  = (const float*)d_in[base + 7];
    P.dpar  = (const float*)d_in[base + 8];  P.ng    = (const float*)d_in[base + 9];
    P.nb    = (const float*)d_in[base + 10]; P.out_w = (const float*)d_in[base + 11];
    P.out_b = (const float*)d_in[base + 12];
    return P;
  };
  SS2DParams sg = getp(5), bm = getp(18);

  float* ws = (float*)d_ws;
  size_t off = 0;
  auto alloc = [&](size_t n) { float* p = ws + off; off += n; return p; };
  float* bins    = alloc((size_t)3 * NBAT * CBIN * LQ);
  float* refined = alloc((size_t)NBAT * CBIN * LQ);
  float* biasb   = alloc((size_t)NBAT * CBIN * LQ);
  float* kld     = alloc((size_t)NBAT * CBIN * LQ);
  float* bnbuf   = alloc((size_t)NBAT * CBIN * LQ);
  float* sigma   = alloc((size_t)NBAT * CBIN * LQ);
  float* xz      = alloc((size_t)NBAT * 96 * LQ);
  float* xconv   = alloc((size_t)NBAT * DINNER * LQ);
  float* xdbl    = alloc((size_t)NBAT * KDIR * 34 * LQ);
  float* dtsb    = alloc((size_t)NBAT * KDIR * DINNER * LQ);
  float* hend    = alloc((size_t)NBAT * KDIR * DINNER * N_CHUNK * NSTATE);
  float* h0b     = alloc((size_t)NBAT * KDIR * DINNER * N_CHUNK * NSTATE);
  float* sdtb    = alloc((size_t)NBAT * KDIR * DINNER * N_CHUNK);
  float* ysb     = alloc((size_t)NBAT * KDIR * DINNER * LQ);
  float* ygb     = alloc((size_t)NBAT * DINNER * LQ);

  k_dec2bin<<<(3 * NBAT * CBIN * LQ + 255) / 256, 256, 0, stream>>>(images, bins);
  k_init<<<(NBAT * CBIN * LQ + 255) / 256, 256, 0, stream>>>(bins, refined, biasb);

  auto run_ss2d = [&](const float* xin, const SS2DParams& P, float* outp) {
    dim3 g1(LQ / 64, NBAT);
    k_gemm<CBIN, 96><<<g1, 128, 0, stream>>>(xin, P.in_w, P.in_b, xz);
    k_dwconv<<<(NBAT * DINNER * LQ) / 256, 256, 0, stream>>>(xz, P.conv_w, P.conv_b, xconv);
    dim3 g2(LQ / 64, NBAT, KDIR);
    k_xdbl<<<g2, 128, 0, stream>>>(xconv, P.xproj, xdbl);
    k_dts<<<(NBAT * KDIR * DINNER * LQ) / 256, 256, 0, stream>>>(xdbl, P.dt_w, P.dt_b, dtsb);
    int units = NBAT * KDIR * DINNER * N_CHUNK;           // 49152
    k_scan_p1<<<(units * 16) / 256, 256, 0, stream>>>(dtsb, xdbl, xconv, P.alog, hend, sdtb);
    k_scan_mid<<<(NBAT * KDIR * DINNER * NSTATE) / 256, 256, 0, stream>>>(hend, sdtb, P.alog, h0b);
    k_scan_p2<<<(units * 16) / 256, 256, 0, stream>>>(dtsb, xdbl, xconv, P.alog, h0b, ysb);
    k_combine<<<(NBAT * LQ) / 256, 256, 0, stream>>>(ysb, xconv, xz, P.dpar, P.ng, P.nb, ygb);
    dim3 g3(LQ / 64, NBAT);
    k_gemm<DINNER, CBIN><<<g3, 128, 0, stream>>>(ygb, P.out_w, P.out_b, outp);
  };

  for (int i = 0; i < 3; i++) {
    k_prep<<<(NBAT * LQ) / 256, 256, 0, stream>>>(bins + (size_t)i * NBAT * CBIN * LQ,
                                                  refined, biasb, nimg_g, nimg_b,
                                                  nbias_g, nbias_b, kld, bnbuf);
    run_ss2d(kld, sg, sigma);
    run_ss2d(bnbuf, bm, biasb);
    k_update<<<(NBAT * CBIN * LQ) / 256, 256, 0, stream>>>(refined, sigma, biasb);
  }
  k_bin2dec<<<(NBAT * 3 * LQ) / 256, 256, 0, stream>>>(refined, (float*)d_out);
}
